// HGTModel_26482768347828
// MI455X (gfx1250) — matrix operand order repaired and checked by reading, not compiled, not run on hardware
//
#include <hip/hip_runtime.h>
#include <hip/hip_bf16.h>
#include <math.h>

// ---------------- problem constants (match reference) ----------------
#define HIDN   64
#define NHEAD  2
#define DHEAD  32
#define NLAYER 2
#define NGRAPH 64
#define FIN    771
#define NAn    120000
#define NEn    60000
#define E0N    1500000
#define E1N    1000000
#define E2N    1000000

#define KTILE  128           // K-tile staged in LDS (4 WMMA chunks of 32)

typedef __attribute__((ext_vector_type(16))) __bf16 v16bf;
typedef __attribute__((ext_vector_type(8)))  float  v8f;
typedef __attribute__((ext_vector_type(4)))  unsigned int v4u;
typedef __attribute__((ext_vector_type(8)))  int v8i;
typedef __attribute__((ext_vector_type(4)))  int v4i;

#if defined(__HIP_DEVICE_COMPILE__) && __has_builtin(__builtin_amdgcn_tensor_load_to_lds) && __has_builtin(__builtin_amdgcn_s_wait_tensorcnt)
#define HAVE_TDM 1
#else
#define HAVE_TDM 0
#endif

union ABFrag { v16bf v; unsigned short u[16]; uint4 q[2]; };

__device__ __forceinline__ unsigned short f2bf(float f) {
  unsigned int b = __float_as_uint(f);
  b += 0x7FFFu + ((b >> 16) & 1u);          // round-to-nearest-even
  return (unsigned short)(b >> 16);
}
__device__ __forceinline__ float geluf(float x) {
  return 0.5f * x * (1.0f + erff(x * 0.70710678118654752f));
}
// order-preserving float<->uint encoding so atomicMax(u32) == segment max(f32)
__device__ __forceinline__ unsigned int encOrd(float f) {
  unsigned int b = __float_as_uint(f);
  return (b & 0x80000000u) ? ~b : (b | 0x80000000u);
}
__device__ __forceinline__ float decOrd(unsigned int u) {
  return (u & 0x80000000u) ? __uint_as_float(u ^ 0x80000000u)
                           : __uint_as_float(~u);
}

// =====================================================================
// Generic GEMM, output width fixed at 64:  C = epilogue(actA(A) @ W + b)
//   A: M x K f32 (lda), W: K x 64 f32 row-major.
//   geluA: exact-GELU on A elements at load (for gelu(agg) @ Wa)
//   Xold/skipP: skip-mix  C = beta*C + (1-beta)*Xold, beta=sigmoid(*skipP)
// 8 waves/block, each wave owns 16 rows x 64 cols (4 WMMA accumulators).
// W K-tile is DMA'd into LDS by the Tensor Data Mover (when available),
// then converted f32->bf16 into WMMA B-fragment order.
// =====================================================================
__global__ __launch_bounds__(256) void gemm_n64_wmma(
    const float* __restrict__ A, int lda,
    const float* __restrict__ Wt, const float* __restrict__ bias,
    float* __restrict__ C, const float* __restrict__ Xold,
    const float* __restrict__ skipP,
    int M, int K, int doRelu, int geluA)
{
  __shared__ unsigned short sB[(KTILE / 32) * 2048]; // bf16 B fragments (16KB)
  __shared__ float sW[KTILE * 64];                   // f32 TDM staging  (32KB)

  const int tid  = threadIdx.x;
  const int wave = tid >> 5;
  const int lane = tid & 31;
  const int half = lane >> 4;               // WMMA half-wave
  const int ln   = lane & 15;
  const long long rowbase = (long long)blockIdx.x * 128 + wave * 16;
  const long long arow    = rowbase + ln;
  const bool rowOK        = arow < (long long)M;
  const bool ldaV4        = (lda & 3) == 0;
  const float* Arow       = A + arow * (long long)lda;

  v8f acc0 = {0,0,0,0,0,0,0,0};
  v8f acc1 = acc0, acc2 = acc0, acc3 = acc0;

  for (int ko = 0; ko < K; ko += KTILE) {
    int nch = (K - ko + 31) >> 5; if (nch > KTILE / 32) nch = KTILE / 32;
#if HAVE_TDM
    // ---- TDM: DMA W[ko:ko+rows, 0:64] (f32, row-major) into LDS ----
    {
      int rows = K - ko; if (rows > KTILE) rows = KTILE;
      if (tid == 0) {
        unsigned lds = (unsigned)(size_t)(void*)sW;
        unsigned long long ga =
            (unsigned long long)(size_t)(const void*)(Wt + (size_t)ko * 64);
        v4u g0; v8i g1;
        v4i g2 = {0, 0, 0, 0}, g3 = {0, 0, 0, 0};
        v8i g4 = {0, 0, 0, 0, 0, 0, 0, 0};
        g0[0] = 1u;                                   // count=1 (valid D#)
        g0[1] = lds;                                  // lds_addr
        g0[2] = (unsigned)(ga & 0xFFFFFFFFu);         // global_addr[31:0]
        g0[3] = (unsigned)((ga >> 32) & 0x01FFFFFFu)  // global_addr[56:32]
              | (2u << 30);                           // type=2 ("image")
        unsigned td0 = 64u, td1 = (unsigned)rows;
        unsigned tile0 = 64u, tile1 = (unsigned)rows;
        unsigned long long str0 = 64ull;              // tensor_dim0_stride
        g1[0] = (int)(2u << 16);                      // data_size = 4B
        g1[1] = (int)((td0 & 0xFFFFu) << 16);         // tensor_dim0 lo16
        g1[2] = (int)(((td0 >> 16) & 0xFFFFu) | ((td1 & 0xFFFFu) << 16));
        g1[3] = (int)(((td1 >> 16) & 0xFFFFu) | ((tile0 & 0xFFFFu) << 16));
        g1[4] = (int)(tile1 & 0xFFFFu);               // tile_dim1, tile_dim2=0
        g1[5] = (int)(unsigned)(str0 & 0xFFFFFFFFu);
        g1[6] = (int)(unsigned)((str0 >> 32) & 0xFFFFu);
        g1[7] = 0;
        // 6-arg form on this toolchain (extra int32x8 group, then cpol)
        __builtin_amdgcn_tensor_load_to_lds(g0, g1, g2, g3, g4, 0);
      }
      __builtin_amdgcn_s_wait_tensorcnt(0);
      __syncthreads();
    }
#endif
    // ---- convert f32 -> bf16 into WMMA B-fragment order ----
    // fragment elem index: (((c*4 + t)*32 + lane)*16 + j)
    //   k = ko + c*32 + (j<8 ? (lane/16)*8 + j : 16 + (lane/16)*8 + (j-8)) ,  n = t*16 + (lane%16)
    for (int i = tid; i < nch * 2048; i += 256) {
      int j   = i & 15;
      int ln2 = (i >> 4) & 31;
      int t   = (i >> 9) & 3;
      int cc  = i >> 11;
      int k   = ko + (cc << 5) + ((j < 8) ? (((ln2 >> 4) << 3) + j) : (16 + ((ln2 >> 4) << 3) + (j - 8)));
      int n   = (t << 4) + (ln2 & 15);
#if HAVE_TDM
      sB[i] = (k < K) ? f2bf(sW[(k - ko) * 64 + n]) : (unsigned short)0;
#else
      sB[i] = (k < K) ? f2bf(Wt[(long long)k * 64 + n]) : (unsigned short)0;
#endif
    }
    __syncthreads();

    for (int c = 0; c < nch; ++c) {
      // ---- A fragment: 16 rows x 32 K, f32 -> bf16 on the fly ----
      int kbase = ko + (c << 5);
      float fa[16];
      if (rowOK && ldaV4 && (kbase + 32 <= K)) {
        // fully in-range chunk: 4x aligned float4 (global_load_b128)
        const float4* p0 = (const float4*)(Arow + kbase + half * 8);
        const float4* p1 = (const float4*)(Arow + kbase + 16 + half * 8);
        float4 f0 = p0[0], f1 = p0[1], f2 = p1[0], f3 = p1[1];
        fa[0]=f0.x; fa[1]=f0.y; fa[2]=f0.z; fa[3]=f0.w;
        fa[4]=f1.x; fa[5]=f1.y; fa[6]=f1.z; fa[7]=f1.w;
        fa[8]=f2.x; fa[9]=f2.y; fa[10]=f2.z; fa[11]=f2.w;
        fa[12]=f3.x; fa[13]=f3.y; fa[14]=f3.z; fa[15]=f3.w;
      } else {
#pragma unroll
        for (int j = 0; j < 16; ++j) {
          int k = kbase + ((j < 8) ? (half * 8 + j) : (16 + half * 8 + (j - 8)));
          fa[j] = (rowOK && k < K) ? Arow[k] : 0.0f;
        }
      }
      ABFrag a;
#pragma unroll
      for (int j = 0; j < 16; ++j)
        a.u[j] = f2bf(geluA ? geluf(fa[j]) : fa[j]);

      // ---- B fragments: 32B contiguous per lane -> ds_load_b128 x2 ----
      const unsigned short* bb = sB + (c << 11) + (lane << 4);
      ABFrag b0, b1, b2, b3;
      b0.q[0] = *(const uint4*)(bb +    0); b0.q[1] = *(const uint4*)(bb +    8);
      b1.q[0] = *(const uint4*)(bb +  512); b1.q[1] = *(const uint4*)(bb +  520);
      b2.q[0] = *(const uint4*)(bb + 1024); b2.q[1] = *(const uint4*)(bb + 1032);
      b3.q[0] = *(const uint4*)(bb + 1536); b3.q[1] = *(const uint4*)(bb + 1544);

      acc0 = __builtin_amdgcn_wmma_f32_16x16x32_bf16(false, a.v, false, b0.v, (short)0, acc0, false, false);
      acc1 = __builtin_amdgcn_wmma_f32_16x16x32_bf16(false, a.v, false, b1.v, (short)0, acc1, false, false);
      acc2 = __builtin_amdgcn_wmma_f32_16x16x32_bf16(false, a.v, false, b2.v, (short)0, acc2, false, false);
      acc3 = __builtin_amdgcn_wmma_f32_16x16x32_bf16(false, a.v, false, b3.v, (short)0, acc3, false, false);
    }
    __syncthreads();
  }

  // ---- epilogue: C/D layout vgpr i -> row half*8+i, col t*16+ln ----
  float beta = 1.0f, omb = 0.0f;
  if (Xold) {
    float s = skipP[0];
    beta = 1.0f / (1.0f + __expf(-s));
    omb  = 1.0f - beta;
  }
#pragma unroll
  for (int t = 0; t < 4; ++t) {
    v8f av = (t == 0) ? acc0 : (t == 1) ? acc1 : (t == 2) ? acc2 : acc3;
    int col  = (t << 4) + ln;
    float bv = bias ? bias[col] : 0.0f;
#pragma unroll
    for (int i = 0; i < 8; ++i) {
      long long m = rowbase + half * 8 + i;
      if (m < (long long)M) {
        float vv = av[i] + bv;
        if (doRelu) vv = fmaxf(vv, 0.0f);
        if (Xold) vv = beta * vv + omb * Xold[m * 64 + col];
        C[m * 64 + col] = vv;
      }
    }
  }
}

// =====================================================================
// Expand per-head DxD relation weight into block-diagonal 64x64 (f32)
// =====================================================================
__global__ void expand_bd(const float* __restrict__ src, float* __restrict__ dst) {
  int i = blockIdx.x * blockDim.x + threadIdx.x;
  if (i >= 64 * 64) return;
  int k = i >> 6, n = i & 63;
  int hk = k >> 5, hn = n >> 5;
  dst[i] = (hk == hn) ? src[hk * DHEAD * DHEAD + (k & 31) * DHEAD + (n & 31)] : 0.0f;
}

// =====================================================================
// Edge phases, one thread per (edge, head). Scores are computed once
// (phase 0) and cached in scBuf; later phases touch 4B instead of 256B.
// =====================================================================
__global__ __launch_bounds__(256) void edge_score(
    const float* __restrict__ Q,  const float* __restrict__ K2,
    const int* __restrict__ src,  const int* __restrict__ dst, int E,
    const float* __restrict__ prelP,
    unsigned int* __restrict__ mEnc, float* __restrict__ scBuf)
{
  long long idx = (long long)blockIdx.x * blockDim.x + threadIdx.x;
  if (idx >= (long long)E * NHEAD) return;
  int e = (int)(idx >> 1);
  int h = (int)(idx & 1);
  int s = src[e], d = dst[e];

  const float* q  = Q  + (long long)d * HIDN + h * DHEAD;
  const float* k2 = K2 + (long long)s * HIDN + h * DHEAD;
  float sc = 0.0f;
#pragma unroll
  for (int j = 0; j < DHEAD; j += 4) {
    float4 qa = *(const float4*)(q + j);
    float4 kb = *(const float4*)(k2 + j);
    sc += qa.x * kb.x + qa.y * kb.y + qa.z * kb.z + qa.w * kb.w;
  }
  sc *= prelP[h] * 0.17677669529663687f;     // prel * 1/sqrt(D)
  scBuf[idx] = sc;
  atomicMax(&mEnc[(unsigned)d * NHEAD + h], encOrd(sc));
}

__global__ __launch_bounds__(256) void edge_norm(
    const int* __restrict__ dst, int E,
    const unsigned int* __restrict__ mEnc, float* __restrict__ den,
    float* __restrict__ scBuf)
{
  long long idx = (long long)blockIdx.x * blockDim.x + threadIdx.x;
  if (idx >= (long long)E * NHEAD) return;
  int e = (int)(idx >> 1);
  int h = (int)(idx & 1);
  int d = dst[e];
  unsigned int mi = (unsigned)d * NHEAD + h;
  float ex = __expf(scBuf[idx] - decOrd(mEnc[mi]));
  scBuf[idx] = ex;                            // cache numerator
  atomicAdd(&den[mi], ex);
}

__global__ __launch_bounds__(256) void edge_agg(
    const float* __restrict__ V2, const int* __restrict__ src,
    const int* __restrict__ dst, int E,
    const float* __restrict__ den, const float* __restrict__ scBuf,
    float* __restrict__ agg)
{
  long long idx = (long long)blockIdx.x * blockDim.x + threadIdx.x;
  if (idx >= (long long)E * NHEAD) return;
  int e = (int)(idx >> 1);
  int h = (int)(idx & 1);
  int s = src[e], d = dst[e];
  float w = scBuf[idx] / den[(unsigned)d * NHEAD + h];
  const float* v2 = V2 + (long long)s * HIDN + h * DHEAD;
  float* ag = agg + (long long)d * HIDN + h * DHEAD;
#pragma unroll
  for (int j = 0; j < DHEAD; j += 4) {
    float4 vv = *(const float4*)(v2 + j);
    atomicAdd(&ag[j + 0], w * vv.x);
    atomicAdd(&ag[j + 1], w * vv.y);
    atomicAdd(&ag[j + 2], w * vv.z);
    atomicAdd(&ag[j + 3], w * vv.w);
  }
}

// ---------------- segment mean (graph pooling) ----------------
__global__ void seg_acc(const float* __restrict__ x, const int* __restrict__ seg,
                        int N, float* __restrict__ sums, float* __restrict__ cnt) {
  int i = blockIdx.x * blockDim.x + threadIdx.x;
  if (i >= N) return;
  int b = seg[i];
  atomicAdd(&cnt[b], 1.0f);
  const float* xr = x + (long long)i * HIDN;
  float* sr = sums + (long long)b * HIDN;
  for (int j = 0; j < HIDN; ++j) atomicAdd(&sr[j], xr[j]);
}
__global__ void seg_div(float* __restrict__ sums, const float* __restrict__ cnt) {
  int i = blockIdx.x * blockDim.x + threadIdx.x;
  if (i >= NGRAPH * HIDN) return;
  sums[i] /= fmaxf(cnt[i >> 6], 1.0f);
}

// ---------------- final head helpers ----------------
__global__ void concat_gh(const float* __restrict__ g, const float* __restrict__ a,
                          float* __restrict__ o) {
  int i = blockIdx.x * blockDim.x + threadIdx.x;
  if (i >= NGRAPH * 128) return;
  int r = i >> 7, c = i & 127;
  o[i] = (c < 64) ? g[r * 64 + c] : a[r * 64 + (c - 64)];
}
__global__ void final_out(const float* __restrict__ h, const float* __restrict__ w,
                          const float* __restrict__ b, float* __restrict__ out) {
  int g = threadIdx.x;
  if (g >= NGRAPH) return;
  float s = b[0];
  for (int j = 0; j < HIDN; ++j) s += h[g * HIDN + j] * w[j];
  out[g] = s;
}

// =====================================================================
// host-side launcher
// =====================================================================
static inline void gemm64(const float* A, int lda, const float* Wt,
                          const float* bias, float* C, const float* Xold,
                          const float* skipP, int M, int K, int relu, int geluA,
                          hipStream_t s) {
  dim3 grid((unsigned)((M + 127) / 128));
  gemm_n64_wmma<<<grid, 256, 0, s>>>(A, lda, Wt, bias, C, Xold, skipP, M, K, relu, geluA);
}

extern "C" void kernel_launch(void* const* d_in, const int* in_sizes, int n_in,
                              void* d_out, int out_size, void* d_ws, size_t ws_size,
                              hipStream_t stream) {
  (void)in_sizes; (void)n_in; (void)out_size; (void)ws_size;

  const float* x_arg = (const float*)d_in[0];
  const float* x_ev  = (const float*)d_in[1];
  const float* Win   = (const float*)d_in[2];
  const float* b_in  = (const float*)d_in[3];
  const float* Wk    = (const float*)d_in[4];
  const float* bk    = (const float*)d_in[5];
  const float* Wq    = (const float*)d_in[6];
  const float* bq    = (const float*)d_in[7];
  const float* Wv    = (const float*)d_in[8];
  const float* bv    = (const float*)d_in[9];
  const float* Wa    = (const float*)d_in[10];
  const float* ba    = (const float*)d_in[11];
  const float* skip  = (const float*)d_in[12];
  const float* Watt  = (const float*)d_in[13];
  const float* Wmsg  = (const float*)d_in[14];
  const float* prel  = (const float*)d_in[15];
  // d_in[16..17] Wnews/bnews, [20..23] Wqm/bqm/Wkm/bkm: dead code in reference
  const float* Wdeb  = (const float*)d_in[18];
  const float* bdeb  = (const float*)d_in[19];
  const float* Wvm   = (const float*)d_in[24];
  const float* bvm   = (const float*)d_in[25];
  const float* Wom   = (const float*)d_in[26];
  const float* bom   = (const float*)d_in[27];
  const float* Wc1   = (const float*)d_in[28];
  const float* bc1   = (const float*)d_in[29];
  const float* Wc2   = (const float*)d_in[30];
  const float* bc2   = (const float*)d_in[31];
  const int* e0s = (const int*)d_in[32];
  const int* e0d = (const int*)d_in[33];
  const int* e1s = (const int*)d_in[34];
  const int* e1d = (const int*)d_in[35];
  const int* e2s = (const int*)d_in[36];
  const int* e2d = (const int*)d_in[37];
  const int* batch = (const int*)d_in[38];

  // ---- workspace bump allocator (float units, 256B-aligned blocks) ----
  float* ws = (float*)d_ws;
  size_t off = 0;
  auto alloc = [&](size_t n) -> float* {
    float* p = ws + off;
    off += (n + 63) & ~(size_t)63;
    return p;
  };
  const size_t NA64 = (size_t)NAn * HIDN, NE64 = (size_t)NEn * HIDN;
  float* x0[2]  = { alloc(NA64), alloc(NA64) };
  float* x1[2]  = { alloc(NE64), alloc(NE64) };
  float* Kb0 = alloc(NA64); float* Qb0 = alloc(NA64); float* Vb0 = alloc(NA64);
  float* Kb1 = alloc(NE64); float* Qb1 = alloc(NE64); float* Vb1 = alloc(NE64);
  float* k2e0 = alloc(NA64); float* v2e0 = alloc(NA64);
  float* k2e1 = alloc(NE64); float* v2e1 = alloc(NE64);
  float* k2e2 = alloc(NA64); float* v2e2 = alloc(NA64);
  float* agg0 = alloc(NA64); float* agg1 = alloc(NE64);
  unsigned int* m0 = (unsigned int*)alloc((size_t)NAn * NHEAD);
  float* den0 = alloc((size_t)NAn * NHEAD);
  unsigned int* m1 = (unsigned int*)alloc((size_t)NEn * NHEAD);
  float* den1 = alloc((size_t)NEn * NHEAD);
  float* sc0  = alloc(((size_t)E0N + E1N) * NHEAD);   // scores, dst-type 0
  float* sc1  = alloc((size_t)E2N * NHEAD);           // scores, dst-type 1
  float* wbd  = alloc(64 * 64);
  float* gemb = alloc(NGRAPH * 64); float* cnt = alloc(NGRAPH);
  float* gbuf = alloc(NGRAPH * 64); float* vbuf = alloc(NGRAPH * 64);
  float* abuf = alloc(NGRAPH * 64); float* gcat = alloc(NGRAPH * 128);
  float* hbuf = alloc(NGRAPH * 64);

  const unsigned g0 = (unsigned)(((long long)E0N * NHEAD + 255) / 256);
  const unsigned g1 = (unsigned)(((long long)E1N * NHEAD + 255) / 256);
  const unsigned g2 = (unsigned)(((long long)E2N * NHEAD + 255) / 256);

  // ---- input projection + ReLU (K=771, streamed once, bf16 WMMA) ----
  gemm64(x_arg, FIN, Win,              b_in,        x0[0], nullptr, nullptr, NAn, FIN, 1, 0, stream);
  gemm64(x_ev,  FIN, Win + FIN * HIDN, b_in + HIDN, x1[0], nullptr, nullptr, NEn, FIN, 1, 0, stream);

  int cur = 0;
  for (int l = 0; l < NLAYER; ++l) {
    // ---- K/Q/V projections per node type ----
    const float* xc[2] = { x0[cur], x1[cur] };
    float* Kt[2] = { Kb0, Kb1 };
    float* Qt[2] = { Qb0, Qb1 };
    float* Vt[2] = { Vb0, Vb1 };
    const int Mt[2] = { NAn, NEn };
    for (int t = 0; t < 2; ++t) {
      int wi = (l * 2 + t);
      gemm64(xc[t], HIDN, Wk + (size_t)wi * HIDN * HIDN, bk + (size_t)wi * HIDN, Kt[t], nullptr, nullptr, Mt[t], HIDN, 0, 0, stream);
      gemm64(xc[t], HIDN, Wq + (size_t)wi * HIDN * HIDN, bq + (size_t)wi * HIDN, Qt[t], nullptr, nullptr, Mt[t], HIDN, 0, 0, stream);
      gemm64(xc[t], HIDN, Wv + (size_t)wi * HIDN * HIDN, bv + (size_t)wi * HIDN, Vt[t], nullptr, nullptr, Mt[t], HIDN, 0, 0, stream);
    }
    // ---- hoisted relation transforms: K2[e]=K@Watt, V2[e]=V@Wmsg ----
    struct { int e, srct; float *k2, *v2; } rel[3] = {
      {0, 0, k2e0, v2e0}, {1, 1, k2e1, v2e1}, {2, 0, k2e2, v2e2}};
    for (int r = 0; r < 3; ++r) {
      size_t wo = (size_t)(l * 3 + rel[r].e) * NHEAD * DHEAD * DHEAD;
      expand_bd<<<16, 256, 0, stream>>>(Watt + wo, wbd);
      gemm64(Kt[rel[r].srct], HIDN, wbd, nullptr, rel[r].k2, nullptr, nullptr, Mt[rel[r].srct], HIDN, 0, 0, stream);
      expand_bd<<<16, 256, 0, stream>>>(Wmsg + wo, wbd);
      gemm64(Vt[rel[r].srct], HIDN, wbd, nullptr, rel[r].v2, nullptr, nullptr, Mt[rel[r].srct], HIDN, 0, 0, stream);
    }
    // ---- segment softmax state ----
    (void)hipMemsetAsync(m0,   0, (size_t)NAn * NHEAD * 4, stream);
    (void)hipMemsetAsync(den0, 0, (size_t)NAn * NHEAD * 4, stream);
    (void)hipMemsetAsync(agg0, 0, NA64 * 4, stream);
    (void)hipMemsetAsync(m1,   0, (size_t)NEn * NHEAD * 4, stream);
    (void)hipMemsetAsync(den1, 0, (size_t)NEn * NHEAD * 4, stream);
    (void)hipMemsetAsync(agg1, 0, NE64 * 4, stream);

    const float* pl = prel + (size_t)l * 3 * NHEAD;
    float* sc_e1 = sc0 + (size_t)E0N * NHEAD;
    // phase 0: scores + segment max
    edge_score<<<g0, 256, 0, stream>>>(Qb0, k2e0, e0s, e0d, E0N, pl + 0, m0, sc0);
    edge_score<<<g1, 256, 0, stream>>>(Qb0, k2e1, e1s, e1d, E1N, pl + 2, m0, sc_e1);
    edge_score<<<g2, 256, 0, stream>>>(Qb1, k2e2, e2s, e2d, E2N, pl + 4, m1, sc1);
    // phase 1: exp + denominator
    edge_norm<<<g0, 256, 0, stream>>>(e0d, E0N, m0, den0, sc0);
    edge_norm<<<g1, 256, 0, stream>>>(e1d, E1N, m0, den0, sc_e1);
    edge_norm<<<g2, 256, 0, stream>>>(e2d, E2N, m1, den1, sc1);
    // phase 2: weighted aggregation
    edge_agg<<<g0, 256, 0, stream>>>(v2e0, e0s, e0d, E0N, den0, sc0, agg0);
    edge_agg<<<g1, 256, 0, stream>>>(v2e1, e1s, e1d, E1N, den0, sc_e1, agg0);
    edge_agg<<<g2, 256, 0, stream>>>(v2e2, e2s, e2d, E2N, den1, sc1, agg1);

    // ---- out projection: beta*(gelu(agg)@Wa + ba) + (1-beta)*x_old ----
    gemm64(agg0, HIDN, Wa + (size_t)(l * 2 + 0) * HIDN * HIDN, ba + (size_t)(l * 2 + 0) * HIDN,
           x0[1 - cur], x0[cur], skip + (l * 2 + 0), NAn, HIDN, 0, 1, stream);
    gemm64(agg1, HIDN, Wa + (size_t)(l * 2 + 1) * HIDN * HIDN, ba + (size_t)(l * 2 + 1) * HIDN,
           x1[1 - cur], x1[cur], skip + (l * 2 + 1), NEn, HIDN, 0, 1, stream);
    cur ^= 1;
  }

  // ---- graph pooling over argument nodes ----
  (void)hipMemsetAsync(gemb, 0, NGRAPH * 64 * 4, stream);
  (void)hipMemsetAsync(cnt,  0, NGRAPH * 4, stream);
  seg_acc<<<(NAn + 255) / 256, 256, 0, stream>>>(x0[cur], batch, NAn, gemb, cnt);
  seg_div<<<(NGRAPH * 64 + 255) / 256, 256, 0, stream>>>(gemb, cnt);

  // ---- head: g = gemb@Wdeb+bdeb ; attn = (g@Wvm+bvm)@Wom+bom ;
  //            h = relu([g,attn]@Wc1+bc1) ; out = h@Wc2+bc2
  gemm64(gemb, HIDN, Wdeb, bdeb, gbuf, nullptr, nullptr, NGRAPH, HIDN, 0, 0, stream);
  gemm64(gbuf, HIDN, Wvm,  bvm,  vbuf, nullptr, nullptr, NGRAPH, HIDN, 0, 0, stream);
  gemm64(vbuf, HIDN, Wom,  bom,  abuf, nullptr, nullptr, NGRAPH, HIDN, 0, 0, stream);
  concat_gh<<<(NGRAPH * 128 + 255) / 256, 256, 0, stream>>>(gbuf, abuf, gcat);
  gemm64(gcat, 128, Wc1, bc1, hbuf, nullptr, nullptr, NGRAPH, 128, 1, 0, stream);
  final_out<<<1, 64, 0, stream>>>(hbuf, Wc2, bc2, (float*)d_out);
}
